// LSTMDiscriminator_80418967650614
// MI455X (gfx1250) — matrix operand order repaired and checked by reading, not compiled, not run on hardware
//
#include <hip/hip_runtime.h>
#include <math.h>

typedef _Float16 v16h __attribute__((ext_vector_type(16)));
typedef _Float16 v8h  __attribute__((ext_vector_type(8)));
typedef float    v8f  __attribute__((ext_vector_type(8)));

#define EDIM 64
#define HDIM 128
#define GDIM 512      // 4*H
#define TILE_M 16
#define MAX_PTS 32    // >= T_obs + T_pred (= 21)

__device__ __forceinline__ float fast_sigmoid(float x) {
  // v_exp_f32 + v_rcp_f32, branch-free, saturates correctly
  return __builtin_amdgcn_rcpf(1.f + __expf(-x));
}
__device__ __forceinline__ float fast_tanh(float x) {
  // tanh(x) = 1 - 2/(1+exp(2x)); exp->inf gives +1, exp->0 gives -1
  return 1.f - 2.f * __builtin_amdgcn_rcpf(1.f + __expf(2.f * x));
}

// ---------------------------------------------------------------------------
// Kernel A: fold W_emb into W_ih (K=64 GEMM collapses to K=2), fuse biases,
// and convert W_hh to f16 for the WMMA path.
// ---------------------------------------------------------------------------
__global__ __launch_bounds__(256) void lstm_prep_kernel(
    const float* __restrict__ W_emb, const float* __restrict__ b_emb,
    const float* __restrict__ W_ih,  const float* __restrict__ W_hh,
    const float* __restrict__ b_ih,  const float* __restrict__ b_hh,
    float* __restrict__ wc, float* __restrict__ biasc,
    _Float16* __restrict__ whh16) {
  int n = blockIdx.x * blockDim.x + threadIdx.x;
  if (n >= GDIM) return;
  float wx = 0.f, wy = 0.f, bb = b_ih[n] + b_hh[n];
  for (int e = 0; e < EDIM; ++e) {
    float w = W_ih[n * EDIM + e];
    wx += w * W_emb[e * 2 + 0];
    wy += w * W_emb[e * 2 + 1];
    bb += w * b_emb[e];
  }
  wc[n * 2 + 0] = wx;
  wc[n * 2 + 1] = wy;
  biasc[n] = bb;
  for (int k = 0; k < HDIM; ++k)
    whh16[n * HDIM + k] = (_Float16)W_hh[n * HDIM + k];
}

// ---------------------------------------------------------------------------
// Kernel B: LSTM recurrence on the gathered "primary" agents + fused MLP.
// Block: 256 threads (8 waves) handles 16 agents. Wave w owns hidden-column
// group jg=w, i.e. gate tiles jg, jg+8, jg+16, jg+24 (= i,f,g,o for the SAME
// hidden columns -> cell update is pure per-lane VALU in C layout).
// W_hh B-fragments and h/c state of the owned columns are register-resident;
// h is additionally mirrored (f16) in a ping-pong LDS buffer as A-frag source.
// ---------------------------------------------------------------------------
__global__ __launch_bounds__(256) void lstm_main_kernel(
    const float* __restrict__ observed, const float* __restrict__ prediction,
    const long long* __restrict__ bsplit,
    const float* __restrict__ wc, const float* __restrict__ biasc,
    const _Float16* __restrict__ whh16,
    const float* __restrict__ W1, const float* __restrict__ b1,
    const float* __restrict__ W2, const float* __restrict__ b2,
    const float* __restrict__ W3, const float* __restrict__ b3,
    float* __restrict__ out, int N, int T_obs, int T_total) {
  __shared__ __align__(32) _Float16 hbuf[2][TILE_M][HDIM];  // ping-pong h state (f16)
  __shared__ float obsbuf[MAX_PTS][TILE_M][2];              // gathered trajectories
  __shared__ float ldswc[GDIM][2];                          // folded K=2 weights
  __shared__ float ldsb[GDIM];                              // fused bias
  __shared__ float hf32[TILE_M][HDIM];                      // final h, f32 (MLP input)
  __shared__ float x1buf[TILE_M][64];
  __shared__ float x2buf[TILE_M][32];

  const int tid  = threadIdx.x;
  const int lane = tid & 31;
  const int wave = tid >> 5;        // = jg
  const int blk  = blockIdx.x;

  // --- gather obs slice for this block's 16 primary agents into LDS ---
  const int npts = T_total + 1;
  for (int p = tid; p < npts * TILE_M * 2; p += blockDim.x) {
    int t = p / (TILE_M * 2);
    int r = p % (TILE_M * 2);
    int m = r >> 1, cmp = r & 1;
    int agent = (int)bsplit[blk * TILE_M + m];
    float v;
    if (t < T_obs) v = observed[((long)t * N + agent) * 2 + cmp];
    else           v = prediction[((long)(t - T_obs) * N + agent) * 2 + cmp];
    obsbuf[t][m][cmp] = v;
  }
  // --- stage folded weights/bias, zero both h buffers ---
  for (int p = tid; p < GDIM; p += blockDim.x) {
    ldswc[p][0] = wc[p * 2 + 0];
    ldswc[p][1] = wc[p * 2 + 1];
    ldsb[p] = biasc[p];
  }
  for (int p = tid; p < 2 * TILE_M * HDIM; p += blockDim.x)
    (&hbuf[0][0][0])[p] = (_Float16)0.f;
  __syncthreads();

  const int nl    = lane & 15;            // N-index within 16x16 tile
  const int hhalf = (lane >> 4) & 1;      // lane group (K/M split)
  const int msel  = hhalf * 8;            // C-layout: M = v + msel
  const int jg    = wave;
  const int jcol  = jg * 16 + nl;         // hidden column owned by this lane

  // gate column indices for i/f/g/o of hidden column jcol
  const int n_i = jcol, n_f = jcol + HDIM, n_g = jcol + 2 * HDIM, n_o = jcol + 3 * HDIM;

  // per-lane constants over all time steps
  const float bias_i = ldsb[n_i], bias_f = ldsb[n_f], bias_g = ldsb[n_g], bias_o = ldsb[n_o];
  const float wix = ldswc[n_i][0], wiy = ldswc[n_i][1];
  const float wfx = ldswc[n_f][0], wfy = ldswc[n_f][1];
  const float wgx = ldswc[n_g][0], wgy = ldswc[n_g][1];
  const float wox = ldswc[n_o][0], woy = ldswc[n_o][1];

  const int aoff  = hhalf * 16;   // A-frag K sub-offset for this lane group
  const int koff0 = hhalf * 16;   // B-frag K sub-offset for this lane group

  // --- hoist W_hh B fragments into registers: 16 x v16h (loop invariant) ---
  v16h bfr_i[4], bfr_f[4], bfr_g[4], bfr_o[4];
  #pragma unroll
  for (int c = 0; c < 4; ++c) {
    int kbase = c * 32 + koff0;
    bfr_i[c] = *(const v16h*)&whh16[n_i * HDIM + kbase];
    bfr_f[c] = *(const v16h*)&whh16[n_f * HDIM + kbase];
    bfr_g[c] = *(const v16h*)&whh16[n_g * HDIM + kbase];
    bfr_o[c] = *(const v16h*)&whh16[n_o * HDIM + kbase];
  }

  // register-resident c and h state for the owned (m, jcol) elements
  float creg[8], hreg[8];
  #pragma unroll
  for (int v = 0; v < 8; ++v) { creg[v] = 0.f; hreg[v] = 0.f; }

  int pp = 0;  // ping-pong selector: read hbuf[pp], write hbuf[pp^1]
  for (int t = 0; t < T_total; ++t) {
    // ---- load A fragments of h (16x128, 4 chunks of K=32) from LDS ----
    v16h afrag[4];
    #pragma unroll
    for (int c = 0; c < 4; ++c) {
      v8h lo = *(const v8h*)&hbuf[pp][nl][c * 32 + aoff];
      v8h hh = *(const v8h*)&hbuf[pp][nl][c * 32 + aoff + 8];
      #pragma unroll
      for (int q = 0; q < 8; ++q) { afrag[c][q] = lo[q]; afrag[c][q + 8] = hh[q]; }
    }
    // ---- per-lane velocity + NaN mask for its 8 agent rows ----
    float vx[8], vy[8];
    bool  msk[8];
    #pragma unroll
    for (int v = 0; v < 8; ++v) {
      int m = v + msel;
      float o1x = obsbuf[t][m][0],     o1y = obsbuf[t][m][1];
      float o2x = obsbuf[t + 1][m][0], o2y = obsbuf[t + 1][m][1];
      bool mk = !((o1x != o1x) || (o2x != o2x));
      msk[v] = mk;
      vx[v] = mk ? (o2x - o1x) * 4.0f : 0.f;   // SCALE folded in
      vy[v] = mk ? (o2y - o1y) * 4.0f : 0.f;
    }
    // ---- init accumulators: fused bias + folded K=2 velocity GEMM ----
    v8f acc_i, acc_f, acc_g, acc_o;
    #pragma unroll
    for (int v = 0; v < 8; ++v) {
      acc_i[v] = bias_i + vx[v] * wix + vy[v] * wiy;
      acc_f[v] = bias_f + vx[v] * wfx + vy[v] * wfy;
      acc_g[v] = bias_g + vx[v] * wgx + vy[v] * wgy;
      acc_o[v] = bias_o + vx[v] * wox + vy[v] * woy;
    }
    // ---- h @ W_hh^T : 4 K-chunks x 4 gate tiles, B register-resident ----
    #pragma unroll
    for (int c = 0; c < 4; ++c) {
      acc_i = __builtin_amdgcn_wmma_f32_16x16x32_f16(false, afrag[c], false, bfr_i[c], (short)0, acc_i, false, false);
      acc_f = __builtin_amdgcn_wmma_f32_16x16x32_f16(false, afrag[c], false, bfr_f[c], (short)0, acc_f, false, false);
      acc_g = __builtin_amdgcn_wmma_f32_16x16x32_f16(false, afrag[c], false, bfr_g[c], (short)0, acc_g, false, false);
      acc_o = __builtin_amdgcn_wmma_f32_16x16x32_f16(false, afrag[c], false, bfr_o[c], (short)0, acc_o, false, false);
    }
    // ---- cell update: branch-free, old h/c carried in registers ----
    #pragma unroll
    for (int v = 0; v < 8; ++v) {
      int m = v + msel;
      float ig = fast_sigmoid(acc_i[v]);
      float fg = fast_sigmoid(acc_f[v]);
      float og = fast_sigmoid(acc_o[v]);
      float gg = fast_tanh(acc_g[v]);
      float cn = fg * creg[v] + ig * gg;
      float hn = og * fast_tanh(cn);
      creg[v] = msk[v] ? cn : creg[v];
      hreg[v] = msk[v] ? hn : hreg[v];
      hbuf[pp ^ 1][m][jcol] = (_Float16)hreg[v];
    }
    __syncthreads();   // writes to hbuf[pp^1] visible before next step's reads
    pp ^= 1;
  }

  // ---- publish final h (f32) for the MLP head ----
  #pragma unroll
  for (int v = 0; v < 8; ++v) hf32[v + msel][jcol] = hreg[v];
  __syncthreads();

  // ---- fused MLP head: 128 ->relu 64 ->relu 32 ->relu 1 ----
  {
    int a = tid & 15, ob = (tid >> 4) * 4;   // 4 outputs per thread
    #pragma unroll
    for (int oo = 0; oo < 4; ++oo) {
      int o = ob + oo;
      float s = b1[o];
      for (int k = 0; k < HDIM; ++k) s += W1[o * HDIM + k] * hf32[a][k];
      x1buf[a][o] = fmaxf(s, 0.f);
    }
  }
  __syncthreads();
  {
    int a = tid & 15, ob = (tid >> 4) * 2;   // 2 outputs per thread
    #pragma unroll
    for (int oo = 0; oo < 2; ++oo) {
      int o = ob + oo;
      float s = b2[o];
      for (int k = 0; k < 64; ++k) s += W2[o * 64 + k] * x1buf[a][k];
      x2buf[a][o] = fmaxf(s, 0.f);
    }
  }
  __syncthreads();
  if (tid < TILE_M) {
    float s = b3[0];
    for (int k = 0; k < 32; ++k) s += W3[k] * x2buf[tid][k];
    out[blk * TILE_M + tid] = fmaxf(s, 0.f);
  }
}

extern "C" void kernel_launch(void* const* d_in, const int* in_sizes, int n_in,
                              void* d_out, int out_size, void* d_ws, size_t ws_size,
                              hipStream_t stream) {
  const float*     observed   = (const float*)d_in[0];
  const float*     prediction = (const float*)d_in[1];
  /* d_in[2] = goals: unused by the reference */
  const long long* bsplit     = (const long long*)d_in[3];
  const float*     W_emb      = (const float*)d_in[4];
  const float*     b_emb      = (const float*)d_in[5];
  const float*     W_ih       = (const float*)d_in[6];
  const float*     W_hh       = (const float*)d_in[7];
  const float*     b_ih       = (const float*)d_in[8];
  const float*     b_hh       = (const float*)d_in[9];
  const float*     W1         = (const float*)d_in[10];
  const float*     b1         = (const float*)d_in[11];
  const float*     W2         = (const float*)d_in[12];
  const float*     b2         = (const float*)d_in[13];
  const float*     W3         = (const float*)d_in[14];
  const float*     b3         = (const float*)d_in[15];
  float* out = (float*)d_out;

  int N      = in_sizes[2] / 2;             // goals is (N, 2)
  int T_obs  = in_sizes[0] / (2 * N);       // 9
  int T_pred = in_sizes[1] / (2 * N);       // 12
  int T_total = T_obs + T_pred - 1;         // 20 steps

  // workspace: [whh16: 512*128 f16][wc: 512*2 f32][biasc: 512 f32]
  _Float16* whh16 = (_Float16*)d_ws;
  float*    wcbuf = (float*)((char*)d_ws + (size_t)GDIM * HDIM * sizeof(_Float16));
  float*    bbuf  = (float*)((char*)wcbuf + (size_t)GDIM * 2 * sizeof(float));

  lstm_prep_kernel<<<(GDIM + 255) / 256, 256, 0, stream>>>(
      W_emb, b_emb, W_ih, W_hh, b_ih, b_hh, wcbuf, bbuf, whh16);

  int nblk = out_size / TILE_M;             // 512/16 = 32 blocks
  lstm_main_kernel<<<nblk, 256, 0, stream>>>(
      observed, prediction, bsplit, wcbuf, bbuf, whh16,
      W1, b1, W2, b2, W3, b3, out, N, T_obs, T_total);
}
